// MyRNN_56186762167075
// MI455X (gfx1250) — matrix operand order repaired and checked by reading, not compiled, not run on hardware
//
#include <hip/hip_runtime.h>
#include <hip/hip_bf16.h>

// ---------------------------------------------------------------------------
// 2-layer SimpleRNN on MI455X (gfx1250), wave32 + WMMA bf16.
//   B=4096, T=80, E=100 (padded 128), U=1024.
// bf16 WMMA 16x16x32 with f32 accum; weights pre-swizzled into B-fragment
// lane order (L2-resident stream); hidden state persistent in LDS.
// 2x2 output-tile register blocking: each A fragment feeds 2 N-tiles, each
// B fragment feeds 2 M-tiles -> 2 memory ops per v_wmma.
// ---------------------------------------------------------------------------

#define RNN_B   4096
#define RNN_T   80
#define RNN_E   100
#define RNN_EP  128          // E padded to multiple of 32 (K for WMMA)
#define RNN_U   1024
#define RNN_V   10000
#define BT      32           // batch rows per block
#define NTHREADS 256         // 8 wave32

typedef __attribute__((ext_vector_type(16))) __bf16 v16bf;
typedef __attribute__((ext_vector_type(8)))  __bf16 v8bf;
typedef __attribute__((ext_vector_type(8)))  float  v8f;

union V16U { v16bf v; v8bf h[2]; };

// ---------------------------------------------------------------------------
// Prep kernel 1: f32 row-major [Kpad x N] weight -> bf16, swizzled into WMMA
// B-fragment order. Tile (nt,kt) = 16 cols x 32 K; within a tile, lane l holds
// its contiguous v16bf fragment:
//   lane<16 : N = nt*16+lane,    elem v -> K = kt*32 + v
//   lane>=16: N = nt*16+lane-16, elem v -> K = kt*32 + 16 + v
// Tiles stored [nt][kt] so the K-loop streams contiguously.
// ---------------------------------------------------------------------------
__global__ void swizzle_weights_bf16(const float* __restrict__ W,
                                     __bf16* __restrict__ dst,
                                     int Kact, int Kpad, int N) {
    int total = Kpad * N;
    int ktiles = Kpad >> 5;
    for (int e = blockIdx.x * blockDim.x + threadIdx.x; e < total;
         e += gridDim.x * blockDim.x) {
        int tile = e >> 9;           // 512 elements per 32x16 tile
        int r    = e & 511;
        int ln   = r >> 4;           // lane 0..31
        int v    = r & 15;           // element within v16bf
        int nt   = tile / ktiles;
        int kt   = tile - nt * ktiles;
        int n    = nt * 16 + (ln & 15);
        int k    = kt * 32 + ((ln >> 4) << 4) + v;
        float val = (k < Kact) ? W[k * N + n] : 0.0f;
        dst[e] = (__bf16)val;
    }
}

// Prep kernel 2: embedding table f32 [V x 100] -> bf16 [V x 128] zero-padded.
__global__ void emb_to_bf16(const float* __restrict__ src,
                            __bf16* __restrict__ dst) {
    int e = blockIdx.x * blockDim.x + threadIdx.x;
    if (e >= RNN_V * RNN_EP) return;
    int row = e >> 7;
    int j   = e & 127;
    dst[e] = (__bf16)((j < RNN_E) ? src[row * RNN_E + j] : 0.0f);
}

// ---------------------------------------------------------------------------
// 2x2-blocked accumulate:
//   c00 += A[m0=0 ]·B[nt0]   c01 += A[m0=0 ]·B[nt0+1]
//   c10 += A[m0=16]·B[nt0]   c11 += A[m0=16]·B[nt0+1]
// A (bf16, LDS), contiguous k order (same as B swizzle):
//   lane<16 : M=lane,    K elems 0..15
//   lane>=16: M=lane-16, K elems 16..31
// B (bf16, global) pre-swizzled: tile (nt,kt) -> 512 contiguous elems,
// lane's fragment at +lane*16.
// ---------------------------------------------------------------------------
__device__ __forceinline__ void accum_gemm_2x2(v8f& c00, v8f& c01, v8f& c10, v8f& c11,
                                               const __bf16* __restrict__ A, int strideA,
                                               const __bf16* __restrict__ Bsw, int ktiles,
                                               int nt0, int lane) {
    const __bf16* ap0 = A + (lane & 15) * strideA + ((lane >> 4) << 4);
    const __bf16* ap1 = ap0 + 16 * strideA;
    const __bf16* bp0 = Bsw + (size_t)(nt0 * ktiles) * 512 + lane * 16;
    const __bf16* bp1 = bp0 + (size_t)ktiles * 512;
    #pragma unroll 2
    for (int kt = 0; kt < ktiles; ++kt) {
        V16U a0, a1, b0, b1;
        a0.h[0] = *(const v8bf*)(ap0 + kt * 32);
        a0.h[1] = *(const v8bf*)(ap0 + kt * 32 + 8);
        a1.h[0] = *(const v8bf*)(ap1 + kt * 32);
        a1.h[1] = *(const v8bf*)(ap1 + kt * 32 + 8);
        b0.h[0] = *(const v8bf*)(bp0 + kt * 512);
        b0.h[1] = *(const v8bf*)(bp0 + kt * 512 + 8);
        b1.h[0] = *(const v8bf*)(bp1 + kt * 512);
        b1.h[1] = *(const v8bf*)(bp1 + kt * 512 + 8);
        c00 = __builtin_amdgcn_wmma_f32_16x16x32_bf16(false, a0.v, false, b0.v, (short)0, c00, false, false);
        c01 = __builtin_amdgcn_wmma_f32_16x16x32_bf16(false, a0.v, false, b1.v, (short)0, c01, false, false);
        c10 = __builtin_amdgcn_wmma_f32_16x16x32_bf16(false, a1.v, false, b0.v, (short)0, c10, false, false);
        c11 = __builtin_amdgcn_wmma_f32_16x16x32_bf16(false, a1.v, false, b1.v, (short)0, c11, false, false);
    }
}

// tanh + bf16 store of one 16x16 f32 accumulator tile into LDS [BT x U].
// C layout: VGPR r -> row m0 + r + (lane>=16 ? 8 : 0), col = col0 + lane%16.
__device__ __forceinline__ void store_tile_tanh(const v8f& c, __bf16* __restrict__ dst,
                                                int m0, int col0, int lane) {
    int col   = col0 + (lane & 15);
    int rbase = m0 + ((lane >> 4) << 3);
    #pragma unroll
    for (int r = 0; r < 8; ++r)
        dst[(rbase + r) * RNN_U + col] = (__bf16)tanhf(c[r]);
}

// One RNN layer for the block's 32xU output: dst = tanh(A1@B1 + A2@B2 + bias).
// 8 waves x 4 iterations x (2x2 tiles) = 128 16x16 tiles.
__device__ __forceinline__ void rnn_layer(const __bf16* __restrict__ A1, int S1,
                                          const __bf16* __restrict__ B1, int Kt1,
                                          const __bf16* __restrict__ A2,
                                          const __bf16* __restrict__ B2,
                                          const float* __restrict__ bias,
                                          __bf16* __restrict__ dst,
                                          int wave, int lane) {
    for (int i = 0; i < 4; ++i) {
        int nt0 = (wave * 4 + i) * 2;   // N-tile pair: cols [nt0*16, nt0*16+32)
        int n0  = nt0 << 4;
        float bv0 = bias[n0 + (lane & 15)];
        float bv1 = bias[n0 + 16 + (lane & 15)];
        v8f c00 = {bv0, bv0, bv0, bv0, bv0, bv0, bv0, bv0};
        v8f c01 = {bv1, bv1, bv1, bv1, bv1, bv1, bv1, bv1};
        v8f c10 = c00;
        v8f c11 = c01;
        accum_gemm_2x2(c00, c01, c10, c11, A1, S1, B1, Kt1, nt0, lane);
        accum_gemm_2x2(c00, c01, c10, c11, A2, RNN_U, B2, RNN_U / 32, nt0, lane);
        store_tile_tanh(c00, dst, 0,  n0,      lane);
        store_tile_tanh(c01, dst, 0,  n0 + 16, lane);
        store_tile_tanh(c10, dst, 16, n0,      lane);
        store_tile_tanh(c11, dst, 16, n0 + 16, lane);
    }
}

// ---------------------------------------------------------------------------
// Persistent RNN kernel: one block owns BT=32 batch rows for all T steps.
// LDS: 3 rotating 32x1024 bf16 h-buffers (192 KB) + 32x128 bf16 x-tile (8 KB).
// ---------------------------------------------------------------------------
__global__ __launch_bounds__(NTHREADS, 1)
void rnn_persistent(const int*    __restrict__ inputs,   // [B,T]
                    const float*  __restrict__ b0,
                    const float*  __restrict__ b1,
                    const float*  __restrict__ Wfc,      // [U]
                    const float*  __restrict__ bfc,      // [1]
                    const __bf16* __restrict__ Wx0s,     // swizzled [128 x U]
                    const __bf16* __restrict__ Wh0s,     // swizzled [U x U]
                    const __bf16* __restrict__ Wx1s,
                    const __bf16* __restrict__ Wh1s,
                    const __bf16* __restrict__ embB,     // [V x 128] bf16
                    float* __restrict__ out) {
    extern __shared__ __align__(16) __bf16 smem[];
    __bf16* h0p = smem;                        // current h0   (32*1024)
    __bf16* h1p = smem + BT * RNN_U;           // current h1
    __bf16* hfp = smem + 2 * BT * RNN_U;       // free / staging
    __bf16* xt  = smem + 3 * BT * RNN_U;       // x tile (32*128)

    const int tid  = threadIdx.x;
    const int lane = tid & 31;
    const int wave = tid >> 5;
    const int brow = blockIdx.x * BT;

    // Zero initial h0/h1 (131072 bytes) with 8B stores.
    {
        unsigned long long* z = (unsigned long long*)smem;
        for (int i = tid; i < (2 * BT * RNN_U) / 4; i += NTHREADS) z[i] = 0ull;
    }
    __syncthreads();

    for (int t = 0; t < RNN_T; ++t) {
        // ---- gather embeddings for this step: 32 rows x 128 bf16 ----
        {
            int row = tid >> 3;           // 32 rows, 8 threads/row
            int seg = (tid & 7) << 4;     // 16 bf16 per thread
            int tok = inputs[(brow + row) * RNN_T + t];
            const __bf16* src = embB + (size_t)tok * RNN_EP + seg;
            __bf16* dstx = xt + row * RNN_EP + seg;
            *(v8bf*)(dstx)     = *(const v8bf*)(src);
            *(v8bf*)(dstx + 8) = *(const v8bf*)(src + 8);
        }
        __syncthreads();

        // ---- layer 0: hf = tanh(xt@Wx0 + h0@Wh0 + b0) ----
        rnn_layer(xt, RNN_EP, Wx0s, RNN_EP / 32, h0p, Wh0s, b0, hfp, wave, lane);
        __syncthreads();

        // ---- layer 1: old-h0 buffer <- tanh(hf@Wx1 + h1@Wh1 + b1) ----
        rnn_layer(hfp, RNN_U, Wx1s, RNN_U / 32, h1p, Wh1s, b1, h0p, wave, lane);
        __syncthreads();

        // rotate: new h0 = hf, new h1 = old h0 slot, free = old h1 slot
        __bf16* nh0 = hfp; __bf16* nh1 = h0p; __bf16* nfr = h1p;
        h0p = nh0; h1p = nh1; hfp = nfr;
    }

    // ---- final dense + sigmoid: out[b] = sigmoid(h1 . Wfc + bfc) ----
    if (tid < BT) {
        const __bf16* h1r = h1p + tid * RNN_U;
        float s = bfc[0];
        for (int j = 0; j < RNN_U; ++j) s += (float)h1r[j] * Wfc[j];
        out[brow + tid] = 1.0f / (1.0f + expf(-s));
    }
}

// ---------------------------------------------------------------------------
// Host launcher
// ---------------------------------------------------------------------------
extern "C" void kernel_launch(void* const* d_in, const int* in_sizes, int n_in,
                              void* d_out, int out_size, void* d_ws, size_t ws_size,
                              hipStream_t stream) {
    const int*   inputs = (const int*)  d_in[0];
    const float* emb    = (const float*)d_in[1];
    const float* Wx0    = (const float*)d_in[2];
    const float* Wh0    = (const float*)d_in[3];
    const float* b0     = (const float*)d_in[4];
    const float* Wx1    = (const float*)d_in[5];
    const float* Wh1    = (const float*)d_in[6];
    const float* b1     = (const float*)d_in[7];
    const float* Wfc    = (const float*)d_in[8];
    const float* bfc    = (const float*)d_in[9];

    // Workspace carve-up (bf16 element offsets).
    __bf16* ws   = (__bf16*)d_ws;
    __bf16* Wx0s = ws;                        // 128*1024  = 131072
    __bf16* Wh0s = ws + 131072;               // 1024*1024 = 1048576
    __bf16* Wx1s = ws + 131072 + 1048576;
    __bf16* Wh1s = ws + 131072 + 2 * 1048576;
    __bf16* embB = ws + 131072 + 3 * 1048576; // 10000*128 = 1280000

    // Weight conversion + swizzle (cheap, once per launch, deterministic).
    swizzle_weights_bf16<<<512,  NTHREADS, 0, stream>>>(Wx0, Wx0s, RNN_E, RNN_EP, RNN_U);
    swizzle_weights_bf16<<<4096, NTHREADS, 0, stream>>>(Wh0, Wh0s, RNN_U, RNN_U, RNN_U);
    swizzle_weights_bf16<<<4096, NTHREADS, 0, stream>>>(Wx1, Wx1s, RNN_U, RNN_U, RNN_U);
    swizzle_weights_bf16<<<4096, NTHREADS, 0, stream>>>(Wh1, Wh1s, RNN_U, RNN_U, RNN_U);
    emb_to_bf16<<<(RNN_V * RNN_EP + NTHREADS - 1) / NTHREADS, NTHREADS, 0, stream>>>(emb, embB);

    // Persistent RNN: 128 blocks x 256 threads, 200 KB dynamic LDS
    // (ISA: up to 320 KB per workgroup on CDNA5).
    const int lds_bytes = (3 * BT * RNN_U + BT * RNN_EP) * (int)sizeof(__bf16);
    hipFuncSetAttribute(reinterpret_cast<const void*>(rnn_persistent),
                        hipFuncAttributeMaxDynamicSharedMemorySize, lds_bytes);
    rnn_persistent<<<RNN_B / BT, NTHREADS, lds_bytes, stream>>>(
        inputs, b0, b1, Wfc, bfc, Wx0s, Wh0s, Wx1s, Wh1s, embB, (float*)d_out);
}
